// LSTMBaseStrategyModelDecoder_26886495273665
// MI455X (gfx1250) — matrix operand order repaired and checked by reading, not compiled, not run on hardware
//
#include <hip/hip_runtime.h>
#include <hip/hip_bf16.h>
#include <math.h>

// ---------------------------------------------------------------------------
// LSTM strategy-model decoder for MI455X (gfx1250), wave32 + WMMA f16.
// All dense GEMMs (LSTM gates, relfeat projection) run through
// v_wmma_f32_16x16x32_f16.  The (B,C,513) order_w tensor of the reference is
// algebraically folded into per-batch vectors so the candidate pass is a pure
// L2-resident gather+dot.
// ---------------------------------------------------------------------------

typedef __attribute__((ext_vector_type(16))) _Float16 v16h;
typedef __attribute__((ext_vector_type(8)))  _Float16 v8h;
typedef __attribute__((ext_vector_type(8)))  float    v8f;

#define B_   64
#define L_   81
#define D2_  512
#define S_   17
#define C_   256
#define V_   13042
#define H_   512
#define E_   80
#define P_   60
#define F_   178
#define KX_  672          // 652 (D2+E+P) padded to a multiple of 32
#define G4H_ 2048         // 4*H
#define BL_  (B_*L_)      // 5184

#define CDIV(a,b) (((a)+(b)-1)/(b))

// ---------------- wave reduction (wave32) ----------------------------------
static __device__ __forceinline__ float wave_sum(float v) {
#pragma unroll
  for (int off = 16; off > 0; off >>= 1) v += __shfl_xor(v, off, 32);
  return v;
}

// ---------------- WMMA fragment loads --------------------------------------
// A (16x32, f16, row-major, stride ld):
//   lanes 0-15  : row m0+lane,    elems 0..15 = K k0..k0+15
//   lanes 16-31 : row m0+lane-16, elems 0..15 = K k0+16..k0+31
static __device__ __forceinline__ v16h load_fragA(const _Float16* __restrict__ p0,
                                                  int ld, int m0, int k0, int lane) {
  int r  = m0 + (lane & 15);
  int kb = k0 + ((lane & 16) ? 16 : 0);
  const _Float16* p = p0 + (size_t)r * ld + kb;
  v8h lo = *reinterpret_cast<const v8h*>(p);
  v8h hi = *reinterpret_cast<const v8h*>(p + 8);
  v16h f;
#pragma unroll
  for (int i = 0; i < 8; i++) { f[i] = lo[i]; f[i + 8] = hi[i]; }
  return f;
}

// B (32x16, f16).  Weight stored as (N,K) row-major (stride ld) so B[k][n] =
// W[n][k].  Contiguous k order, same as load_fragA:
//   lanes 0-15  : col n0+lane,   elems 0..15 = K k0..k0+15
//   lanes 16-31 : col n0+lane-16, elems 0..15 = K k0+16..k0+31
static __device__ __forceinline__ v16h load_fragB(const _Float16* __restrict__ p0,
                                                  int ld, int n0, int k0, int lane) {
  int n  = n0 + (lane & 15);
  int kb = k0 + ((lane & 16) ? 16 : 0);
  const _Float16* p = p0 + (size_t)n * ld + kb;
  v8h lo = *reinterpret_cast<const v8h*>(p);
  v8h hi = *reinterpret_cast<const v8h*>(p + 8);
  v16h f;
#pragma unroll
  for (int i = 0; i < 8; i++) { f[i] = lo[i]; f[i + 8] = hi[i]; }
  return f;
}

// ---------------- generic WMMA GEMM:  C = A*B^T (+ A2*B2^T) + bias ----------
// A: M x K f16 (lda), Bw: N x K f16 (ldb).  One wave per 16x16 C tile.
__global__ void wmma_gemm_kernel(const _Float16* __restrict__ A,  int lda,
                                 const _Float16* __restrict__ Bw, int ldb,
                                 const _Float16* __restrict__ A2, int lda2,
                                 const _Float16* __restrict__ B2, int ldb2,
                                 const float* __restrict__ bias0,
                                 const float* __restrict__ bias1,
                                 float* __restrict__ C, int ldc,
                                 int Mtiles, int Ntiles, int K, int K2) {
  int wave = (blockIdx.x * blockDim.x + threadIdx.x) >> 5;
  int lane = threadIdx.x & 31;
  if (wave >= Mtiles * Ntiles) return;            // wave-uniform: EXEC stays all-1s
  int m0 = (wave / Ntiles) << 4;
  int n0 = (wave % Ntiles) << 4;

  v8f acc = {0.f, 0.f, 0.f, 0.f, 0.f, 0.f, 0.f, 0.f};
  for (int k0 = 0; k0 < K; k0 += 32) {
    v16h af = load_fragA(A, lda, m0, k0, lane);
    v16h bf = load_fragB(Bw, ldb, n0, k0, lane);
    acc = __builtin_amdgcn_wmma_f32_16x16x32_f16(false, af, false, bf,
                                                 (short)0, acc, false, false);
  }
  if (A2) {
    for (int k0 = 0; k0 < K2; k0 += 32) {
      v16h af = load_fragA(A2, lda2, m0, k0, lane);
      v16h bf = load_fragB(B2, ldb2, n0, k0, lane);
      acc = __builtin_amdgcn_wmma_f32_16x16x32_f16(false, af, false, bf,
                                                   (short)0, acc, false, false);
    }
  }
  // C/D layout: VGPR r, lanes 0-15 -> D[r][lane]; lanes 16-31 -> D[8+r][lane-16]
  int ln = lane & 15;
  int mh = (lane & 16) ? 8 : 0;
  float bn = 0.f;
  if (bias0) bn += bias0[n0 + ln];
  if (bias1) bn += bias1[n0 + ln];
#pragma unroll
  for (int r = 0; r < 8; r++)
    C[(size_t)(m0 + mh + r) * ldc + n0 + ln] = acc[r] + bn;
}

// ---------------- conversion / precompute kernels ---------------------------
__global__ void cvt_f32_f16_pad(const float* __restrict__ src, _Float16* __restrict__ dst,
                                int rows, int kin, int kout) {
  int i = blockIdx.x * blockDim.x + threadIdx.x;
  if (i >= rows * kout) return;
  int r = i / kout, k = i - r * kout;
  dst[i] = (k < kin) ? (_Float16)src[(size_t)r * kin + k] : (_Float16)0.f;
}

// relfeat weight (D2=512 x 513) -> transposed f16 (512 cols x 512 k), col 512 dropped
__global__ void cvt_transpose_relw(const float* __restrict__ wsrc, _Float16* __restrict__ wt) {
  int i = blockIdx.x * blockDim.x + threadIdx.x;
  if (i >= 512 * 512) return;
  int n = i >> 9, k = i & 511;
  wt[i] = (_Float16)wsrc[(size_t)k * 513 + n];
}

// scalar "513th column" of relfeat:  relB[b,l] = enc[b,l]·W[:,512] + b[512]
__global__ void relb_kernel(const float* __restrict__ enc,
                            const float* __restrict__ wsrc, const float* __restrict__ bs,
                            const float* __restrict__ wdst, const float* __restrict__ bd,
                            float* __restrict__ rbS, float* __restrict__ rbD) {
  int wv = (blockIdx.x * blockDim.x + threadIdx.x) >> 5;
  int lane = threadIdx.x & 31;
  if (wv >= BL_) return;
  float ps = 0.f, pd = 0.f;
  const float* e = enc + (size_t)wv * 512;
  for (int k = lane; k < 512; k += 32) {
    float ev = e[k];
    ps += ev * wsrc[(size_t)k * 513 + 512];
    pd += ev * wdst[(size_t)k * 513 + 512];
  }
  ps = wave_sum(ps); pd = wave_sum(pd);
  if (lane == 0) { rbS[wv] = ps + bs[512]; rbD[wv] = pd + bd[512]; }
}

// loc_enc_all[s,b,:] = normalize(mask@MA) @ enc[b]
__global__ void loc_enc_kernel(const float* __restrict__ enc, const int* __restrict__ loc_idxs,
                               const float* __restrict__ MA, float* __restrict__ locenc) {
  int s = blockIdx.x / B_, b = blockIdx.x % B_;
  __shared__ float msk[L_];
  __shared__ float al[L_];
  __shared__ float ssum;
  int t = threadIdx.x;
  if (t < L_) {
    int li = loc_idxs[b * L_ + t];
    msk[t] = (li == s || li == -2) ? 1.f : 0.f;
  }
  __syncthreads();
  if (t < L_) {
    float a = 0.f;
    for (int l = 0; l < L_; l++) a += msk[l] * MA[l * L_ + t];
    al[t] = a;
  }
  __syncthreads();
  if (t == 0) {
    float sm = 0.f;
    for (int m = 0; m < L_; m++) sm += al[m];
    ssum = sm + 1e-5f;
  }
  __syncthreads();
  float inv = 1.f / ssum;
  for (int d = t; d < 512; d += blockDim.x) {
    float acc = 0.f;
    for (int m = 0; m < L_; m++) acc += al[m] * enc[((size_t)b * L_ + m) * 512 + d];
    locenc[((size_t)s * B_ + b) * 512 + d] = acc * inv;
  }
}

__global__ void powemb_kernel(const int* __restrict__ power, const float* __restrict__ plw,
                              const float* __restrict__ plb, float* __restrict__ pe) {
  int i = blockIdx.x * blockDim.x + threadIdx.x;
  if (i >= B_ * S_ * P_) return;
  int p = i % P_;
  int bs = i / P_;                      // b*S + s
  pe[i] = plw[power[bs] * P_ + p] + plb[p];
}

__global__ void teacher_copy_kernel(const int* __restrict__ tch, float* __restrict__ out) {
  int i = blockIdx.x * blockDim.x + threadIdx.x;
  if (i < B_ * S_) out[i] = (float)tch[i];
}

// ---------------- per-step kernels ------------------------------------------
// x_f16 = [loc_enc(512) | order_emb(80) | power_emb(60) | zero pad to 672]
__global__ void xasm_kernel(const float* __restrict__ locenc_t, const float* __restrict__ oemb,
                            const float* __restrict__ powemb, int t, _Float16* __restrict__ xh) {
  int i = blockIdx.x * blockDim.x + threadIdx.x;
  if (i >= B_ * KX_) return;
  int b = i / KX_, col = i - b * KX_;
  float v = 0.f;
  if (col < 512)       v = locenc_t[b * 512 + col];
  else if (col < 592)  v = oemb[b * E_ + (col - 512)];
  else if (col < 652)  v = powemb[(b * S_ + t) * P_ + (col - 592)];
  xh[i] = (_Float16)v;
}

__global__ void lstm_act_kernel(const float* __restrict__ gates, float* __restrict__ c,
                                float* __restrict__ h, _Float16* __restrict__ hh) {
  int i = blockIdx.x * blockDim.x + threadIdx.x;
  if (i >= B_ * H_) return;
  int b = i >> 9, j = i & 511;
  const float* g = gates + (size_t)b * G4H_;
  float ig = g[j], fg = g[512 + j], gg = g[1024 + j], og = g[1536 + j];
  float si = 1.f / (1.f + expf(-ig));
  float sf = 1.f / (1.f + expf(-fg));
  float so = 1.f / (1.f + expf(-og));
  float cn = sf * c[i] + si * tanhf(gg);
  float hn = so * tanhf(cn);
  c[i] = cn; h[i] = hn; hh[i] = (_Float16)hn;
}

// per-batch: v[f]=W_w@out+W_b, s=b_w·out+b_b, e_src/e_dst = emb_bias·out1
__global__ void outprep_kernel(const float* __restrict__ out,
                               const float* __restrict__ dww, const float* __restrict__ dwb,
                               const float* __restrict__ dbw, const float* __restrict__ dbb,
                               const float* __restrict__ ebs, const float* __restrict__ ebd,
                               float* __restrict__ v, float* __restrict__ sbuf,
                               float* __restrict__ esrc, float* __restrict__ edst) {
  int b = blockIdx.x;
  __shared__ float sh[H_];
  for (int k = threadIdx.x; k < H_; k += blockDim.x) sh[k] = out[(size_t)b * H_ + k];
  __syncthreads();
  int t = threadIdx.x;
  if (t < F_) {
    float a = 0.f;
    for (int hx = 0; hx < H_; hx++) a += dww[(size_t)t * H_ + hx] * sh[hx];
    v[b * F_ + t] = a + dbw[t];
  } else if (t == F_) {
    float a = 0.f;
    for (int hx = 0; hx < H_; hx++) a += dwb[hx] * sh[hx];
    sbuf[b] = a + dbb[0];
  } else if (t == F_ + 1) {
    float a = 0.f;
    for (int hx = 0; hx < H_; hx++) a += ebs[hx] * sh[hx];
    esrc[b] = a + ebs[512];
  } else if (t == F_ + 2) {
    float a = 0.f;
    for (int hx = 0; hx < H_; hx++) a += ebd[hx] * sh[hx];
    edst[b] = a + ebd[512];
  }
}

// per (b,l): dot of relfeat row with out (+ scalar column term)
__global__ void dots_kernel(const float* __restrict__ relS, const float* __restrict__ relD,
                            const float* __restrict__ rbS, const float* __restrict__ rbD,
                            const float* __restrict__ out,
                            float* __restrict__ dS, float* __restrict__ dD) {
  int wv = (blockIdx.x * blockDim.x + threadIdx.x) >> 5;
  int lane = threadIdx.x & 31;
  if (wv >= BL_) return;
  int b = wv / L_;
  const float* o = out + (size_t)b * H_;
  float ps = 0.f, pd = 0.f;
  for (int k = lane; k < H_; k += 32) {
    float ov = o[k];
    ps += relS[(size_t)wv * H_ + k] * ov;
    pd += relD[(size_t)wv * H_ + k] * ov;
  }
  ps = wave_sum(ps); pd = wave_sum(pd);
  if (lane == 0) { dS[wv] = ps + rbS[wv]; dD[wv] = pd + rbD[wv]; }
}

// candidate scoring: one wave per (b,c)
__global__ void logits_kernel(const int* __restrict__ cands,
                              const int* __restrict__ osrc, const int* __restrict__ odst,
                              const float* __restrict__ cemb, const float* __restrict__ ofeat,
                              const float* __restrict__ out,  const float* __restrict__ v,
                              const float* __restrict__ sbuf, const float* __restrict__ esrc,
                              const float* __restrict__ edst, const float* __restrict__ dS,
                              const float* __restrict__ dD, int t,
                              float* __restrict__ out_logits) {
  int b  = blockIdx.x >> 5;            // 32 blocks of 8 candidates each per batch
  int cg = blockIdx.x & 31;
  int wv = threadIdx.x >> 5, lane = threadIdx.x & 31;
  int c  = cg * 8 + wv;
  int cand = cands[((size_t)b * S_ + t) * C_ + c];
  int sc = cand < 0 ? 0 : cand;
  const float* o = out + (size_t)b * H_;
  float p1 = 0.f;
  for (int k = lane; k < H_; k += 32) p1 += cemb[(size_t)sc * H_ + k] * o[k];
  float p2 = 0.f;
  for (int k = lane; k < F_; k += 32) p2 += ofeat[(size_t)sc * F_ + k] * v[b * F_ + k];
  float tot = wave_sum(p1 + p2);
  if (lane == 0) {
    float lg = tot + sbuf[b];
    int src = osrc[sc], dst = odst[sc];
    if (src > 0) { int s2 = src > L_ - 1 ? L_ - 1 : src; lg += dS[b * L_ + s2] + esrc[b]; }
    if (dst > 0) { int d2 = dst > L_ - 1 ? L_ - 1 : dst; lg += dD[b * L_ + d2] + edst[b]; }
    out_logits[((size_t)b * S_ + t) * C_ + c] = (cand != -1) ? lg : -1e9f;
  }
}

__global__ void local_kernel(const int* __restrict__ cands, const int* __restrict__ tch,
                             int t, float* __restrict__ out_local) {
  int b = blockIdx.x;
  __shared__ int smin;
  if (threadIdx.x == 0) smin = C_;
  __syncthreads();
  int c = threadIdx.x;
  if (cands[((size_t)b * S_ + t) * C_ + c] == tch[b * S_ + t]) atomicMin(&smin, c);
  __syncthreads();
  if (threadIdx.x == 0) out_local[b * S_ + t] = (float)(smin == C_ ? 0 : smin);
}

// new order_emb for next step from teacher order
__global__ void oemb_kernel(const int* __restrict__ tch, int t,
                            const float* __restrict__ oembW, const float* __restrict__ ofeat,
                            const float* __restrict__ oflw, const float* __restrict__ oflb,
                            float* __restrict__ oembbuf) {
  int b = blockIdx.x, e = threadIdx.x;
  if (e >= E_) return;
  int tc = tch[b * S_ + t];
  float a = oembW[(size_t)tc * E_ + e] + oflb[e];
  for (int f = 0; f < F_; f++) a += ofeat[(size_t)tc * F_ + f] * oflw[f * E_ + e];
  oembbuf[b * E_ + e] = a;
}

// ---------------------------------------------------------------------------
extern "C" void kernel_launch(void* const* d_in, const int* in_sizes, int n_in,
                              void* d_out, int out_size, void* d_ws, size_t ws_size,
                              hipStream_t stream) {
  (void)in_sizes; (void)n_in; (void)out_size;
  const float* enc   = (const float*)d_in[0];
  const int*   locix = (const int*)d_in[1];
  const int*   cands = (const int*)d_in[2];
  const int*   power = (const int*)d_in[3];
  const int*   tch   = (const int*)d_in[4];
  const float* MA    = (const float*)d_in[5];
  const float* oembW = (const float*)d_in[6];
  const float* cembW = (const float*)d_in[7];
  const float* plw   = (const float*)d_in[8];
  const float* plb   = (const float*)d_in[9];
  const float* wih0  = (const float*)d_in[10];
  const float* whh0  = (const float*)d_in[11];
  const float* bih0  = (const float*)d_in[12];
  const float* bhh0  = (const float*)d_in[13];
  const float* wih1  = (const float*)d_in[14];
  const float* whh1  = (const float*)d_in[15];
  const float* bih1  = (const float*)d_in[16];
  const float* bhh1  = (const float*)d_in[17];
  const float* ofeat = (const float*)d_in[18];
  const float* oflw  = (const float*)d_in[19];
  const float* oflb  = (const float*)d_in[20];
  const float* dww   = (const float*)d_in[21];
  const float* dwb   = (const float*)d_in[22];
  const float* dbw   = (const float*)d_in[23];
  const float* dbb   = (const float*)d_in[24];
  const float* rsw   = (const float*)d_in[25];
  const float* rsb   = (const float*)d_in[26];
  const float* rdw   = (const float*)d_in[27];
  const float* rdb   = (const float*)d_in[28];
  // d_in[29], d_in[31] (emb_relfeat_*_w) multiply a zero tensor -> unused
  const float* ebs   = (const float*)d_in[30];
  const float* ebd   = (const float*)d_in[32];
  const int*   osrc  = (const int*)d_in[33];
  const int*   odst  = (const int*)d_in[34];

  // ---- workspace layout (all 256B aligned) ----
  char* w = (char*)d_ws;
  size_t off = 0;
  auto take = [&](size_t bytes) -> void* {
    void* p = w + off;
    off = (off + bytes + 255) & ~(size_t)255;
    return p;
  };
  // zero-initialized state block (contiguous, memset once per call)
  float*     H0    = (float*)take((size_t)B_ * H_ * 4);
  float*     H1    = (float*)take((size_t)B_ * H_ * 4);
  float*     C0    = (float*)take((size_t)B_ * H_ * 4);
  float*     C1    = (float*)take((size_t)B_ * H_ * 4);
  float*     OEMB  = (float*)take((size_t)B_ * E_ * 4);
  _Float16*  H0H   = (_Float16*)take((size_t)B_ * H_ * 2);
  _Float16*  H1H   = (_Float16*)take((size_t)B_ * H_ * 2);
  size_t stateBytes = off;
  // f16 weights / activations
  _Float16* WIH0H = (_Float16*)take((size_t)G4H_ * KX_ * 2);
  _Float16* WHH0H = (_Float16*)take((size_t)G4H_ * H_ * 2);
  _Float16* WIH1H = (_Float16*)take((size_t)G4H_ * H_ * 2);
  _Float16* WHH1H = (_Float16*)take((size_t)G4H_ * H_ * 2);
  _Float16* WSRCT = (_Float16*)take((size_t)512 * 512 * 2);
  _Float16* WDSTT = (_Float16*)take((size_t)512 * 512 * 2);
  _Float16* ENCH  = (_Float16*)take((size_t)BL_ * 512 * 2);
  float*    RELS  = (float*)take((size_t)BL_ * 512 * 4);
  float*    RELD  = (float*)take((size_t)BL_ * 512 * 4);
  float*    RELSB = (float*)take((size_t)BL_ * 4);
  float*    RELDB = (float*)take((size_t)BL_ * 4);
  float*    LOCE  = (float*)take((size_t)S_ * B_ * 512 * 4);
  float*    POWE  = (float*)take((size_t)B_ * S_ * P_ * 4);
  _Float16* XH    = (_Float16*)take((size_t)B_ * KX_ * 2);
  float*    GATES = (float*)take((size_t)B_ * G4H_ * 4);
  float*    VBUF  = (float*)take((size_t)B_ * F_ * 4);
  float*    SBUF  = (float*)take((size_t)B_ * 4);
  float*    ESRC  = (float*)take((size_t)B_ * 4);
  float*    EDST  = (float*)take((size_t)B_ * 4);
  float*    DOTS  = (float*)take((size_t)BL_ * 4);
  float*    DOTD  = (float*)take((size_t)BL_ * 4);
  if (off > ws_size) return;   // insufficient scratch

  float* out_teacher = (float*)d_out;              // (B,S)
  float* out_local   = out_teacher + B_ * S_;      // (B,S)
  float* out_logits  = out_local + B_ * S_;        // (B,S,C)

  hipMemsetAsync(w, 0, stateBytes, stream);

  // ---- one-time conversions ----
  cvt_f32_f16_pad<<<CDIV(G4H_ * KX_, 256), 256, 0, stream>>>(wih0, WIH0H, G4H_, 652, KX_);
  cvt_f32_f16_pad<<<CDIV(G4H_ * H_, 256), 256, 0, stream>>>(whh0, WHH0H, G4H_, H_, H_);
  cvt_f32_f16_pad<<<CDIV(G4H_ * H_, 256), 256, 0, stream>>>(wih1, WIH1H, G4H_, H_, H_);
  cvt_f32_f16_pad<<<CDIV(G4H_ * H_, 256), 256, 0, stream>>>(whh1, WHH1H, G4H_, H_, H_);
  cvt_f32_f16_pad<<<CDIV(BL_ * 512, 256), 256, 0, stream>>>(enc, ENCH, BL_, 512, 512);
  cvt_transpose_relw<<<CDIV(512 * 512, 256), 256, 0, stream>>>(rsw, WSRCT);
  cvt_transpose_relw<<<CDIV(512 * 512, 256), 256, 0, stream>>>(rdw, WDSTT);

  // ---- precompute: relfeat GEMMs (5184x512x512 each, WMMA) ----
  wmma_gemm_kernel<<<CDIV((BL_ / 16) * 32 * 32, 256), 256, 0, stream>>>(
      ENCH, 512, WSRCT, 512, nullptr, 0, nullptr, 0, rsb, nullptr,
      RELS, 512, BL_ / 16, 32, 512, 0);
  wmma_gemm_kernel<<<CDIV((BL_ / 16) * 32 * 32, 256), 256, 0, stream>>>(
      ENCH, 512, WDSTT, 512, nullptr, 0, nullptr, 0, rdb, nullptr,
      RELD, 512, BL_ / 16, 32, 512, 0);
  relb_kernel<<<CDIV(BL_ * 32, 256), 256, 0, stream>>>(enc, rsw, rsb, rdw, rdb, RELSB, RELDB);

  loc_enc_kernel<<<S_ * B_, 128, 0, stream>>>(enc, locix, MA, LOCE);
  powemb_kernel<<<CDIV(B_ * S_ * P_, 256), 256, 0, stream>>>(power, plw, plb, POWE);
  teacher_copy_kernel<<<CDIV(B_ * S_, 256), 256, 0, stream>>>(tch, out_teacher);

  // ---- sequential 17-step scan ----
  for (int t = 0; t < S_; t++) {
    xasm_kernel<<<CDIV(B_ * KX_, 256), 256, 0, stream>>>(
        LOCE + (size_t)t * B_ * 512, OEMB, POWE, t, XH);

    // cell 0: gates = x@Wih0^T + h0@Whh0^T + biases   (64x2048, K=672+512)
    wmma_gemm_kernel<<<(4 * 128) / 8, 256, 0, stream>>>(
        XH, KX_, WIH0H, KX_, H0H, H_, WHH0H, H_, bih0, bhh0,
        GATES, G4H_, 4, 128, KX_, H_);
    lstm_act_kernel<<<CDIV(B_ * H_, 256), 256, 0, stream>>>(GATES, C0, H0, H0H);

    // cell 1: gates = h0@Wih1^T + h1@Whh1^T + biases
    wmma_gemm_kernel<<<(4 * 128) / 8, 256, 0, stream>>>(
        H0H, H_, WIH1H, H_, H1H, H_, WHH1H, H_, bih1, bhh1,
        GATES, G4H_, 4, 128, H_, H_);
    lstm_act_kernel<<<CDIV(B_ * H_, 256), 256, 0, stream>>>(GATES, C1, H1, H1H);

    // candidate-pass folded vectors + per-(b,l) relfeat dots
    outprep_kernel<<<B_, 256, 0, stream>>>(H1, dww, dwb, dbw, dbb, ebs, ebd,
                                           VBUF, SBUF, ESRC, EDST);
    dots_kernel<<<CDIV(BL_ * 32, 256), 256, 0, stream>>>(RELS, RELD, RELSB, RELDB,
                                                         H1, DOTS, DOTD);
    logits_kernel<<<B_ * 32, 256, 0, stream>>>(cands, osrc, odst, cembW, ofeat, H1,
                                               VBUF, SBUF, ESRC, EDST, DOTS, DOTD,
                                               t, out_logits);
    local_kernel<<<B_, 256, 0, stream>>>(cands, tch, t, out_local);
    oemb_kernel<<<B_, 128, 0, stream>>>(tch, t, oembW, ofeat, oflw, oflb, OEMB);
  }
}
